// Attention_23072564314402
// MI455X (gfx1250) — matrix operand order repaired and checked by reading, not compiled, not run on hardware
//
#include <hip/hip_runtime.h>

// ---------------------------------------------------------------------------
// Problem geometry (fixed by the reference)
// ---------------------------------------------------------------------------
#define BATCH   2
#define SEQ     2048
#define DIM     1024
#define NHEADS  16
#define NKV     4
#define HDIM    64
#define MTOT    (BATCH * SEQ)        // 4096 rows in all projections

typedef __attribute__((ext_vector_type(16))) __bf16 v16bf;
typedef __attribute__((ext_vector_type(8)))  float  v8f;

#define WMMA_BF16(A, B, C) \
  __builtin_amdgcn_wmma_f32_16x16x32_bf16(false, (A), false, (B), (short)0, (C), false, false)

// ---------------------------------------------------------------------------
// Fragment loaders (ISA 7.12.2 layouts, wave32)
// A-frag (16x32, MxK): lane -> m = lane&15, kh = lane>>4
//   a[i]   (i<8)  = A[m][k0 + 8*kh + i]
//   a[8+i] (i<8)  = A[m][k0 + 16 + 8*kh + i]
// B-frag (32x16, KxN): lane -> n = lane&15, kh = lane>>4
//   b[i]   (i<8)  = Bsrc[n][k0 + 8*kh + i], b[8+i] = Bsrc[n][k0 + 16 + 8*kh + i]   (Bsrc row-major [N][K])
// C-frag (16x16 f32): lane -> n = lane&15; c[r] -> m = r + 8*(lane>>4)
// ---------------------------------------------------------------------------
__device__ inline v16bf load_a_frag(const __bf16* tile, int ld, int k0) {
  const int lane = threadIdx.x & 31;
  const int m = lane & 15, kh = lane >> 4;
  const __bf16* p = tile + (size_t)m * ld + k0 + 8 * kh;
  v16bf a;
#pragma unroll
  for (int i = 0; i < 8; ++i) { a[i] = p[i]; a[8 + i] = p[16 + i]; }
  return a;
}

__device__ inline v16bf load_b_frag(const __bf16* tile, int ld, int k0) {
  const int lane = threadIdx.x & 31;
  const int n = lane & 15, kh = lane >> 4;
  const __bf16* p = tile + (size_t)n * ld + k0 + 8 * kh;
  v16bf b;
#pragma unroll
  for (int i = 0; i < 8; ++i) { b[i] = p[i]; b[8 + i] = p[16 + i]; }
  return b;
}

__device__ inline v8f v8f_zero() {
  v8f z = {0.f, 0.f, 0.f, 0.f, 0.f, 0.f, 0.f, 0.f};
  return z;
}

__device__ inline float half16_max(float v) {
#pragma unroll
  for (int off = 1; off < 16; off <<= 1) v = fmaxf(v, __shfl_xor(v, off, 32));
  return v;
}
__device__ inline float half16_sum(float v) {
#pragma unroll
  for (int off = 1; off < 16; off <<= 1) v += __shfl_xor(v, off, 32);
  return v;
}

// Pin inner-loop schedule: keep VMEM reads in flight behind WMMAs.
__device__ inline void sched_pipeline_hint() {
#if __has_builtin(__builtin_amdgcn_sched_group_barrier)
  __builtin_amdgcn_sched_group_barrier(0x020, 6, 0);   // 6 VMEM reads
#pragma unroll
  for (int t = 0; t < 8; ++t) {
    __builtin_amdgcn_sched_group_barrier(0x008, 1, 0); // 1 WMMA
    __builtin_amdgcn_sched_group_barrier(0x020, 2, 0); // 2 VMEM reads
  }
#endif
}

// ---------------------------------------------------------------------------
// 0) fp32 -> bf16 conversion (4 elements / thread)
// ---------------------------------------------------------------------------
__global__ void cvt_f32_bf16(const float* __restrict__ in, __bf16* __restrict__ out, int n4) {
  int i = blockIdx.x * blockDim.x + threadIdx.x;
  if (i >= n4) return;
  float4 v = ((const float4*)in)[i];
  __bf16* o = out + (size_t)i * 4;
  o[0] = (__bf16)v.x; o[1] = (__bf16)v.y; o[2] = (__bf16)v.z; o[3] = (__bf16)v.w;
}

// ---------------------------------------------------------------------------
// 1) Generic bf16 WMMA GEMM:  C[m][n] = sum_k A[m][k] * W[n][k]
//    A: [M][K] bf16, W: [N][K] bf16, C: [M][N] f32.
//    block = 128 threads (4 waves); block tile 64(M) x 64(N); wave tile 16x64.
//    K-loop unrolled x2 with independent even/odd fragments; schedule pinned
//    so loads stay in flight behind WMMAs.
// ---------------------------------------------------------------------------
__global__ __launch_bounds__(128)
void gemm_bf16_wmma(const __bf16* __restrict__ A,
                    const __bf16* __restrict__ W,
                    float* __restrict__ C,
                    int M, int N, int K) {
  const int wave = threadIdx.x >> 5;
  const int lane = threadIdx.x & 31;
  const int m0 = blockIdx.y * 64 + wave * 16;
  const int n0 = blockIdx.x * 64;

  v8f acc[4];
#pragma unroll
  for (int j = 0; j < 4; ++j) acc[j] = v8f_zero();

  const __bf16* Atile = A + (size_t)m0 * K;

  for (int k0 = 0; k0 < K; k0 += 64) {
    // prefetch one 64-step ahead (lowers to global_prefetch_b8)
    if (k0 + 64 < K) {
      __builtin_prefetch(Atile + (size_t)(lane & 15) * K + k0 + 64, 0, 1);
      __builtin_prefetch(W + (size_t)(n0 + lane) * K + k0 + 64, 0, 1);
      __builtin_prefetch(W + (size_t)(n0 + 32 + lane) * K + k0 + 64, 0, 1);
    }
    // issue ALL loads for both K-substeps first ...
    v16bf ae = load_a_frag(Atile, K, k0);
    v16bf be[4];
#pragma unroll
    for (int j = 0; j < 4; ++j) be[j] = load_b_frag(W + (size_t)(n0 + j * 16) * K, K, k0);
    v16bf ao = load_a_frag(Atile, K, k0 + 32);
    v16bf bo[4];
#pragma unroll
    for (int j = 0; j < 4; ++j) bo[j] = load_b_frag(W + (size_t)(n0 + j * 16) * K, K, k0 + 32);
    // ... then 8 WMMAs (later loads in flight during earlier WMMAs)
#pragma unroll
    for (int j = 0; j < 4; ++j) acc[j] = WMMA_BF16(ae, be[j], acc[j]);
#pragma unroll
    for (int j = 0; j < 4; ++j) acc[j] = WMMA_BF16(ao, bo[j], acc[j]);
    sched_pipeline_hint();
  }

  const int mrow = m0 + (lane >> 4) * 8;
  const int ncol = n0 + (lane & 15);
#pragma unroll
  for (int j = 0; j < 4; ++j)
#pragma unroll
    for (int r = 0; r < 8; ++r)
      C[(size_t)(mrow + r) * N + ncol + j * 16] = acc[j][r];
}

// ---------------------------------------------------------------------------
// 2) RoPE + convert + head-split:  raw f32 [M][nh*64] -> bf16 [b][nh][s][64]
// ---------------------------------------------------------------------------
__global__ void rope_pack(const float* __restrict__ raw,
                          const float* __restrict__ cosT,
                          const float* __restrict__ sinT,
                          __bf16* __restrict__ out, int nh, int total) {
  int tid = blockIdx.x * blockDim.x + threadIdx.x;
  if (tid >= total) return;                  // total = MTOT * nh * 32
  int i = tid & 31;
  int h = (tid >> 5) % nh;
  int m = tid / (32 * nh);
  int s = m & (SEQ - 1);
  int b = m >> 11;
  size_t src = (size_t)m * (nh * HDIM) + h * HDIM + 2 * i;
  float x0 = raw[src], x1 = raw[src + 1];
  float c0 = cosT[s * HDIM + 2 * i], c1 = cosT[s * HDIM + 2 * i + 1];
  float s0 = sinT[s * HDIM + 2 * i], s1 = sinT[s * HDIM + 2 * i + 1];
  size_t dst = (((size_t)(b * nh + h)) * SEQ + s) * HDIM + 2 * i;
  out[dst]     = (__bf16)(x0 * c0 - x1 * s0);
  out[dst + 1] = (__bf16)(x1 * c1 + x0 * s1);
}

// ---------------------------------------------------------------------------
// 3) V pack: raw f32 [M][NKV*64] -> bf16 transposed [b][kv][64][S]
// ---------------------------------------------------------------------------
__global__ void v_pack(const float* __restrict__ raw, __bf16* __restrict__ out, int total) {
  int tid = blockIdx.x * blockDim.x + threadIdx.x;
  if (tid >= total) return;                  // total = MTOT * NKV*HDIM
  int col = tid & (NKV * HDIM - 1);
  int m = tid >> 8;
  int d = col & (HDIM - 1);
  int kv = col >> 6;
  int s = m & (SEQ - 1);
  int b = m >> 11;
  out[(((size_t)(b * NKV + kv)) * HDIM + d) * SEQ + s] = (__bf16)raw[(size_t)m * (NKV * HDIM) + col];
}

// ---------------------------------------------------------------------------
// 4) Causal flash attention, bf16 WMMA, online softmax.
//    grid = (SEQ/64, BATCH*NHEADS); block = 128 (4 waves); wave owns 16 q-rows.
//    Q: [b][h][s][64], K: [b][kv][s][64], Vt: [b][kv][64][s]
//    out attn: bf16 [b][s][h*64] row-major ([4096][1024]).
// ---------------------------------------------------------------------------
__global__ __launch_bounds__(128)
void flash_attn_wmma(const __bf16* __restrict__ Qb,
                     const __bf16* __restrict__ Kb,
                     const __bf16* __restrict__ Vt,
                     __bf16* __restrict__ attn) {
  __shared__ __bf16 Plds[4][16][40];         // per-wave 16x32 P tile (+pad)

  const int wave = threadIdx.x >> 5;
  const int lane = threadIdx.x & 31;
  const int n    = lane & 15;
  const int half = lane >> 4;

  const int bh  = blockIdx.y;
  const int b   = bh >> 4;
  const int h   = bh & 15;
  const int kvh = h >> 2;                    // N_REP = 4

  const int q0 = blockIdx.x * 64 + wave * 16;

  const __bf16* Qh = Qb + ((size_t)(b * NHEADS + h)) * SEQ * HDIM;
  const __bf16* Kh = Kb + ((size_t)(b * NKV + kvh)) * SEQ * HDIM;
  const __bf16* Vh = Vt + ((size_t)(b * NKV + kvh)) * HDIM * SEQ;

  const v16bf qa0 = load_a_frag(Qh + (size_t)q0 * HDIM, HDIM, 0);
  const v16bf qa1 = load_a_frag(Qh + (size_t)q0 * HDIM, HDIM, 32);

  const float NEG = -1.0e30f;
  const float scale = 0.125f;                // 1/sqrt(64)

  float mrow[8], lrow[8];
  v8f oacc[4];
#pragma unroll
  for (int r = 0; r < 8; ++r) { mrow[r] = NEG; lrow[r] = 0.f; }
#pragma unroll
  for (int j = 0; j < 4; ++j) oacc[j] = v8f_zero();

  for (int k0 = 0; k0 < q0 + 16; k0 += 32) {
    // prefetch next K tile rows (one 128B row per lane)
    if (k0 + 32 < q0 + 16)
      __builtin_prefetch(Kh + (size_t)(k0 + 32 + lane) * HDIM, 0, 1);

    // ---- V B-frags first: consumed last (after softmax), longest in flight
    v16bf bv[4];
#pragma unroll
    for (int j = 0; j < 4; ++j) bv[j] = load_b_frag(Vh + (size_t)(j * 16) * SEQ, SEQ, k0);

    // ---- scores: two 16x16 tiles covering keys [k0, k0+32) ----
    float sc[2][8];
#pragma unroll
    for (int s2 = 0; s2 < 2; ++s2) {
      const int kb = k0 + s2 * 16;
      v16bf bk0 = load_b_frag(Kh + (size_t)kb * HDIM, HDIM, 0);
      v16bf bk1 = load_b_frag(Kh + (size_t)kb * HDIM, HDIM, 32);
      v8f t = WMMA_BF16(qa0, bk0, v8f_zero());
      t = WMMA_BF16(qa1, bk1, t);
#pragma unroll
      for (int r = 0; r < 8; ++r) {
        int qrow = q0 + 8 * half + r;
        int key  = kb + n;
        sc[s2][r] = (key <= qrow) ? t[r] * scale : NEG;
      }
    }

    // ---- online softmax ----
    float scf[8];
#pragma unroll
    for (int r = 0; r < 8; ++r) {
      float tmax = half16_max(fmaxf(sc[0][r], sc[1][r]));
      float mnew = fmaxf(mrow[r], tmax);
      scf[r] = __expf(mrow[r] - mnew);
      float p0 = __expf(sc[0][r] - mnew);
      float p1 = __expf(sc[1][r] - mnew);
      Plds[wave][8 * half + r][n]      = (__bf16)p0;
      Plds[wave][8 * half + r][16 + n] = (__bf16)p1;
      float rsum = half16_sum(p0 + p1);
      lrow[r] = lrow[r] * scf[r] + rsum;
      mrow[r] = mnew;
    }
#pragma unroll
    for (int j = 0; j < 4; ++j)
#pragma unroll
      for (int r = 0; r < 8; ++r) oacc[j][r] *= scf[r];

    __builtin_amdgcn_wave_barrier();         // order LDS writes vs reads (wave-sync)

    // ---- P (C-layout) -> A-frag via LDS ----
    v16bf pa;
    {
      const int m = lane & 15, kh = lane >> 4;
#pragma unroll
      for (int i = 0; i < 8; ++i) {
        pa[i]     = Plds[wave][m][8 * kh + i];
        pa[8 + i] = Plds[wave][m][16 + 8 * kh + i];
      }
    }
    __builtin_amdgcn_wave_barrier();

    // ---- O += P @ V ----
#pragma unroll
    for (int j = 0; j < 4; ++j) oacc[j] = WMMA_BF16(pa, bv[j], oacc[j]);
  }

  // ---- epilogue: normalize + store bf16 [b][s][h*64+d] ----
#pragma unroll
  for (int j = 0; j < 4; ++j)
#pragma unroll
    for (int r = 0; r < 8; ++r) {
      int qrow = q0 + 8 * half + r;
      float v = oacc[j][r] / lrow[r];
      attn[((size_t)(b * SEQ + qrow)) * DIM + h * HDIM + j * 16 + n] = (__bf16)v;
    }
}

// ---------------------------------------------------------------------------
// Host launcher
// ---------------------------------------------------------------------------
extern "C" void kernel_launch(void* const* d_in, const int* in_sizes, int n_in,
                              void* d_out, int out_size, void* d_ws, size_t ws_size,
                              hipStream_t stream) {
  const float* X    = (const float*)d_in[0];   // [2][2048][1024]
  const float* fcos = (const float*)d_in[1];   // [2048][64]
  const float* fsin = (const float*)d_in[2];   // [2048][64]
  const float* Wq   = (const float*)d_in[3];   // [1024][1024]
  const float* Wk   = (const float*)d_in[4];   // [256][1024]
  const float* Wv   = (const float*)d_in[5];   // [256][1024]
  const float* Wo   = (const float*)d_in[6];   // [1024][1024]
  float* out = (float*)d_out;

  // workspace layout (bytes), all 256B aligned
  char* ws = (char*)d_ws;
  size_t off = 0;
  auto carve = [&](size_t bytes) { char* p = ws + off; off += (bytes + 255) & ~(size_t)255; return p; };
  __bf16* Xb   = (__bf16*)carve((size_t)MTOT * DIM * 2);          // 8 MB
  __bf16* Wqb  = (__bf16*)carve((size_t)DIM * DIM * 2);           // 2 MB
  __bf16* Wkb  = (__bf16*)carve((size_t)NKV * HDIM * DIM * 2);    // 0.5 MB
  __bf16* Wvb  = (__bf16*)carve((size_t)NKV * HDIM * DIM * 2);    // 0.5 MB
  __bf16* Wob  = (__bf16*)carve((size_t)DIM * DIM * 2);           // 2 MB
  float*  Qraw = (float*)carve((size_t)MTOT * DIM * 4);           // 16 MB
  float*  Kraw = (float*)carve((size_t)MTOT * NKV * HDIM * 4);    // 4 MB
  float*  Vraw = (float*)carve((size_t)MTOT * NKV * HDIM * 4);    // 4 MB
  __bf16* Qbh  = (__bf16*)carve((size_t)MTOT * DIM * 2);          // 8 MB
  __bf16* Kbh  = (__bf16*)carve((size_t)MTOT * NKV * HDIM * 2);   // 2 MB
  __bf16* Vth  = (__bf16*)carve((size_t)MTOT * NKV * HDIM * 2);   // 2 MB
  __bf16* Attn = (__bf16*)carve((size_t)MTOT * DIM * 2);          // 8 MB
  (void)ws_size; (void)n_in; (void)in_sizes; (void)out_size;

  // 0) convert inputs to bf16
  auto cvt = [&](const float* src, __bf16* dst, int n) {
    int n4 = n / 4;
    cvt_f32_bf16<<<(n4 + 255) / 256, 256, 0, stream>>>(src, dst, n4);
  };
  cvt(X,  Xb,  MTOT * DIM);
  cvt(Wq, Wqb, DIM * DIM);
  cvt(Wk, Wkb, NKV * HDIM * DIM);
  cvt(Wv, Wvb, NKV * HDIM * DIM);
  cvt(Wo, Wob, DIM * DIM);

  // 1) QKV projections (WMMA GEMMs); block tile 64x64, 128 threads
  gemm_bf16_wmma<<<dim3(DIM / 64, MTOT / 64), 128, 0, stream>>>(Xb, Wqb, Qraw, MTOT, DIM, DIM);
  gemm_bf16_wmma<<<dim3((NKV * HDIM) / 64, MTOT / 64), 128, 0, stream>>>(Xb, Wkb, Kraw, MTOT, NKV * HDIM, DIM);
  gemm_bf16_wmma<<<dim3((NKV * HDIM) / 64, MTOT / 64), 128, 0, stream>>>(Xb, Wvb, Vraw, MTOT, NKV * HDIM, DIM);

  // 2) RoPE + repack
  {
    int nq = MTOT * NHEADS * (HDIM / 2);
    rope_pack<<<(nq + 255) / 256, 256, 0, stream>>>(Qraw, fcos, fsin, Qbh, NHEADS, nq);
    int nk = MTOT * NKV * (HDIM / 2);
    rope_pack<<<(nk + 255) / 256, 256, 0, stream>>>(Kraw, fcos, fsin, Kbh, NKV, nk);
    int nv = MTOT * NKV * HDIM;
    v_pack<<<(nv + 255) / 256, 256, 0, stream>>>(Vraw, Vth, nv);
  }

  // 3) causal flash attention
  flash_attn_wmma<<<dim3(SEQ / 64, BATCH * NHEADS), 128, 0, stream>>>(Qbh, Kbh, Vth, Attn);

  // 4) output projection -> fp32 d_out
  gemm_bf16_wmma<<<dim3(DIM / 64, MTOT / 64), 128, 0, stream>>>(Attn, Wob, out, MTOT, DIM, DIM);
}
